// TeacherGNN_14542759264928
// MI455X (gfx1250) — matrix operand order repaired and checked by reading, not compiled, not run on hardware
//
#include <hip/hip_runtime.h>

typedef __bf16 v16bf __attribute__((ext_vector_type(16)));
typedef float  v8f   __attribute__((ext_vector_type(8)));
typedef unsigned v8u __attribute__((ext_vector_type(8)));

// ---------------------------------------------------------------- degrees
__global__ __launch_bounds__(256) void gcn_deg_init(float* __restrict__ deg, int n) {
  int i = blockIdx.x * 256 + threadIdx.x;
  if (i < n) deg[i] = 1.0f;  // self loop
}

__global__ __launch_bounds__(256) void gcn_deg_count(float* __restrict__ deg,
                                                     const int* __restrict__ dst, int e) {
  int i = blockIdx.x * 256 + threadIdx.x;
  if (i < e)
    __hip_atomic_fetch_add(&deg[dst[i]], 1.0f, __ATOMIC_RELAXED, __HIP_MEMORY_SCOPE_AGENT);
}

__global__ __launch_bounds__(256) void gcn_rsqrt(float* __restrict__ dinv, int n) {
  int i = blockIdx.x * 256 + threadIdx.x;
  if (i < n) dinv[i] = rsqrtf(dinv[i]);  // deg >= 1 always
}

// ---------------------------------------------------------------- GEMM + fused self-loop epilogue
// Out[N,NCOL] = A[N,128] @ W[128,NCOL];  Agg[N,NCOL] = dinv[row]^2 * Out  (self-loop init)
// bf16 in / f32 accumulate, v_wmma_f32_16x16x32_bf16. 4 waves/block, 16 rows/wave.
template <int NCOL>
__global__ __launch_bounds__(128) void gcn_gemm_bf16(const float* __restrict__ A,
                                                     const float* __restrict__ W,
                                                     const float* __restrict__ dinv,
                                                     float* __restrict__ Out,
                                                     float* __restrict__ Agg, int nrows) {
  constexpr int K      = 128;
  constexpr int NPAIR  = K / 2;   // 64 k-pairs
  constexpr int STRIDE = 68;      // padded pair-stride (16B aligned, bank-friendly)
  __shared__ unsigned w_lds[NCOL * STRIDE];

  const int tid = threadIdx.x;
  // Stage W as packed bf16x2, layout [n][p]: word(n,p) = {bf16 W[2p][n], bf16 W[2p+1][n]}
  for (int i = tid; i < NCOL * NPAIR; i += 128) {
    int p = i / NCOL, n = i % NCOL;               // consecutive tid -> consecutive n (coalesced)
    __bf16 lo = (__bf16)W[(2 * p) * NCOL + n];
    __bf16 hi = (__bf16)W[(2 * p + 1) * NCOL + n];
    typedef __bf16 bf2 __attribute__((ext_vector_type(2)));
    bf2 pk = {lo, hi};
    w_lds[n * STRIDE + p] = __builtin_bit_cast(unsigned, pk);
  }
  __syncthreads();

  const int wave  = tid >> 5;
  const int lane  = tid & 31;
  const int half  = lane >> 4;
  const int mlane = lane & 15;
  const int row_base = blockIdx.x * 64 + wave * 16;

  // A fragments (ISA 16-bit A 16x32 layout): per 32-wide K chunk, lane reads two
  // contiguous 8-float runs: [c*32 + half*8 .. +7] and [c*32 + 16 + half*8 .. +7].
  v16bf afrag[4];
  {
    int row = row_base + mlane;
    if (row >= nrows) row = nrows - 1;  // clamp loads; EXEC must stay all-1s for WMMA
    const float4* ap = (const float4*)(A + (size_t)row * K);
#pragma unroll
    for (int c = 0; c < 4; ++c) {
      float4 r0 = ap[c * 8 + half * 2];
      float4 r1 = ap[c * 8 + half * 2 + 1];
      float4 r2 = ap[c * 8 + half * 2 + 4];
      float4 r3 = ap[c * 8 + half * 2 + 5];
      v16bf a;
      a[0]  = (__bf16)r0.x; a[1]  = (__bf16)r0.y; a[2]  = (__bf16)r0.z; a[3]  = (__bf16)r0.w;
      a[4]  = (__bf16)r1.x; a[5]  = (__bf16)r1.y; a[6]  = (__bf16)r1.z; a[7]  = (__bf16)r1.w;
      a[8]  = (__bf16)r2.x; a[9]  = (__bf16)r2.y; a[10] = (__bf16)r2.z; a[11] = (__bf16)r2.w;
      a[12] = (__bf16)r3.x; a[13] = (__bf16)r3.y; a[14] = (__bf16)r3.z; a[15] = (__bf16)r3.w;
      afrag[c] = a;
    }
  }

  // self-loop weights per output row (same for every n-tile)
  float w2[8];
#pragma unroll
  for (int r = 0; r < 8; ++r) {
    int row = row_base + half * 8 + r;
    float dv = dinv[row < nrows ? row : (nrows - 1)];
    w2[r] = dv * dv;
  }
  const bool full = (row_base + 16) <= nrows;

#pragma unroll
  for (int nt = 0; nt < NCOL / 16; ++nt) {
    const int ncol = nt * 16 + mlane;
    v8f acc = {};
#pragma unroll
    for (int c = 0; c < 4; ++c) {
      // B fragment: 4 packed words at pair index c*16 + half*4 and 4 at c*16 + 8 + half*4 of column ncol
      const unsigned* bp = &w_lds[ncol * STRIDE + c * 16 + half * 4];
      uint4 q0 = *(const uint4*)(bp);
      uint4 q1 = *(const uint4*)(bp + 8);
      v8u bw = {q0.x, q0.y, q0.z, q0.w, q1.x, q1.y, q1.z, q1.w};
      v16bf b = __builtin_bit_cast(v16bf, bw);
      acc = __builtin_amdgcn_wmma_f32_16x16x32_bf16(false, afrag[c], false, b,
                                                    (short)0, acc, false, false);
    }
    // C/D layout: VGPR r -> row (half*8 + r), lane -> column mlane
    if (full) {
#pragma unroll
      for (int r = 0; r < 8; ++r) {
        size_t off = (size_t)(row_base + half * 8 + r) * NCOL + ncol;
        Out[off] = acc[r];
        Agg[off] = w2[r] * acc[r];
      }
    } else {
#pragma unroll
      for (int r = 0; r < 8; ++r) {
        int row = row_base + half * 8 + r;
        if (row < nrows) {
          size_t off = (size_t)row * NCOL + ncol;
          Out[off] = acc[r];
          Agg[off] = w2[r] * acc[r];
        }
      }
    }
  }
}

// ---------------------------------------------------------------- edge aggregation
// agg[dst] += dinv[src]*dinv[dst] * h[src]; one wave covers one edge (F=128) -> coalesced 512B
template <int F>
__global__ __launch_bounds__(256) void gcn_edge_agg(const float* __restrict__ h,
                                                    float* __restrict__ agg,
                                                    const int* __restrict__ src,
                                                    const int* __restrict__ dst,
                                                    const float* __restrict__ dinv,
                                                    int e) {
  constexpr int G = F / 4;
  long long t = (long long)blockIdx.x * 256 + threadIdx.x;
  if (t >= (long long)e * G) return;
  int ei = (int)(t / G), g = (int)(t % G);
  int s = src[ei], d = dst[ei];
  float norm = dinv[s] * dinv[d];
  float4 v = *(const float4*)(h + (size_t)s * F + g * 4);
  float* p = agg + (size_t)d * F + g * 4;
  __hip_atomic_fetch_add(p + 0, norm * v.x, __ATOMIC_RELAXED, __HIP_MEMORY_SCOPE_AGENT);
  __hip_atomic_fetch_add(p + 1, norm * v.y, __ATOMIC_RELAXED, __HIP_MEMORY_SCOPE_AGENT);
  __hip_atomic_fetch_add(p + 2, norm * v.z, __ATOMIC_RELAXED, __HIP_MEMORY_SCOPE_AGENT);
  __hip_atomic_fetch_add(p + 3, norm * v.w, __ATOMIC_RELAXED, __HIP_MEMORY_SCOPE_AGENT);
}

template <int F>
__global__ __launch_bounds__(256) void gcn_bias_relu(float* __restrict__ y,
                                                     const float* __restrict__ b, int n) {
  long long t = (long long)blockIdx.x * 256 + threadIdx.x;
  if (t >= (long long)n * F) return;
  float v = y[t] + b[(int)(t % F)];
  y[t] = v > 0.0f ? v : 0.0f;
}

// fused layer-2 finalize: out[i] = bout + sum_c relu(agg2[i][c] + b2[c]) * Wout[c]
__global__ __launch_bounds__(256) void gcn_finalize(const float* __restrict__ agg2,
                                                    const float* __restrict__ b2,
                                                    const float* __restrict__ Wout,
                                                    const float* __restrict__ bout,
                                                    float* __restrict__ out, int n) {
  int i = blockIdx.x * 256 + threadIdx.x;
  if (i >= n) return;
  float s = bout[0];
  const float4* yp = (const float4*)(agg2 + (size_t)i * 64);
  const float4* bp = (const float4*)b2;
  const float4* wp = (const float4*)Wout;
#pragma unroll
  for (int g = 0; g < 16; ++g) {
    float4 a = yp[g], b = bp[g], w = wp[g];
    float v0 = a.x + b.x, v1 = a.y + b.y, v2 = a.z + b.z, v3 = a.w + b.w;
    v0 = v0 > 0.f ? v0 : 0.f; v1 = v1 > 0.f ? v1 : 0.f;
    v2 = v2 > 0.f ? v2 : 0.f; v3 = v3 > 0.f ? v3 : 0.f;
    s += v0 * w.x + v1 * w.y + v2 * w.z + v3 * w.w;
  }
  out[i] = s;
}

// ---------------------------------------------------------------- launch
static inline unsigned cdiv(long long a, long long b) { return (unsigned)((a + b - 1) / b); }

extern "C" void kernel_launch(void* const* d_in, const int* in_sizes, int n_in,
                              void* d_out, int out_size, void* d_ws, size_t ws_size,
                              hipStream_t stream) {
  const float* x    = (const float*)d_in[0];
  const int*   ei   = (const int*)  d_in[1];
  const float* W1   = (const float*)d_in[2];
  const float* b1   = (const float*)d_in[3];
  const float* W2   = (const float*)d_in[4];
  const float* b2   = (const float*)d_in[5];
  const float* Wout = (const float*)d_in[6];
  const float* bout = (const float*)d_in[7];
  float* out = (float*)d_out;

  const int N = in_sizes[0] / 128;
  const int E = in_sizes[1] / 2;
  const int* src = ei;
  const int* dst = ei + E;

  // workspace layout (floats): dinv[N] | buf1[N*128] | buf2[N*128]  (~103 MB)
  float* ws   = (float*)d_ws;
  size_t nAl  = ((size_t)N + 3) & ~(size_t)3;     // keep 16B alignment for float4
  float* dinv = ws;
  float* buf1 = ws + nAl;
  float* buf2 = buf1 + (size_t)N * 128;

  // degrees -> dinv
  gcn_deg_init <<<cdiv(N, 256), 256, 0, stream>>>(dinv, N);
  gcn_deg_count<<<cdiv(E, 256), 256, 0, stream>>>(dinv, dst, E);
  gcn_rsqrt    <<<cdiv(N, 256), 256, 0, stream>>>(dinv, N);

  // layer 1: h1 = x @ W1 -> buf1, agg1 (self-loop init fused) -> buf2
  gcn_gemm_bf16<128><<<cdiv(N, 64), 128, 0, stream>>>(x, W1, dinv, buf1, buf2, N);
  gcn_edge_agg<128><<<cdiv((long long)E * 32, 256), 256, 0, stream>>>(buf1, buf2, src, dst, dinv, E);
  gcn_bias_relu<128><<<cdiv((long long)N * 128, 256), 256, 0, stream>>>(buf2, b1, N);

  // layer 2: h2 = y1 @ W2 -> buf1[0:N*64], agg2 (fused init) -> buf1[N*64:2*N*64]
  float* h2   = buf1;
  float* agg2 = buf1 + (size_t)N * 64;
  gcn_gemm_bf16<64><<<cdiv(N, 64), 128, 0, stream>>>(buf2, W2, dinv, h2, agg2, N);
  gcn_edge_agg<64><<<cdiv((long long)E * 16, 256), 256, 0, stream>>>(h2, agg2, src, dst, dinv, E);

  // fused bias + relu + readout
  gcn_finalize<<<cdiv(N, 256), 256, 0, stream>>>(agg2, b2, Wout, bout, out, N);
}
